// Block_46119358824793
// MI455X (gfx1250) — matrix operand order repaired and checked by reading, not compiled, not run on hardware
//
#include <hip/hip_runtime.h>
#include <hip/hip_bf16.h>

// ---- problem constants (match reference) ----
#define BATCH 2
#define SEQ   2048
#define CH    768
#define HEADS 12
#define HDIM  64
#define HID_  3072
#define MROWS (BATCH * SEQ)   // 4096
#define ATTN_SCALE 0.125f     // 64^-0.5

typedef __attribute__((ext_vector_type(16))) __bf16 bf16x16;
typedef __attribute__((ext_vector_type(8)))  float  f32x8;

union FragU { bf16x16 v; uint4 u[2]; };

// A-operand (16x32, MxK): lane L<16 holds row L, K chunks {0..7} and {16..23};
// lane L>=16 holds row L-16, K chunks {8..15} and {24..31}.
__device__ inline bf16x16 load_frag_a(const __hip_bfloat16* base, int ld, int row0, int k0, int lane) {
  int r = lane & 15, h = lane >> 4;
  const __hip_bfloat16* p = base + (size_t)(row0 + r) * ld + k0 + h * 8;
  FragU f;
  f.u[0] = *reinterpret_cast<const uint4*>(p);
  f.u[1] = *reinterpret_cast<const uint4*>(p + 16);
  return f.v;
}

// B-operand (32x16, KxN) sourced from an NxK row-major matrix (i.e. computes X @ W^T):
// lane L<16 holds column (row of W) L, K chunks {0..7} and {16..23}; lane L>=16 holds K chunks {8..15} and {24..31}.
__device__ inline bf16x16 load_frag_b(const __hip_bfloat16* base, int ld, int n0, int k0, int lane) {
  int n = lane & 15, h = lane >> 4;
  const __hip_bfloat16* p = base + (size_t)(n0 + n) * ld + k0 + h * 8;
  FragU f;
  f.u[0] = *reinterpret_cast<const uint4*>(p);
  f.u[1] = *reinterpret_cast<const uint4*>(p + 16);
  return f.v;
}

// A-operand loaded from a 16x32 bf16 row-major LDS tile (stride 32 elements).
__device__ inline bf16x16 load_frag_a_lds(const __hip_bfloat16* pl, int lane) {
  int r = lane & 15, h = lane >> 4;
  FragU f;
  f.u[0] = *reinterpret_cast<const uint4*>(pl + r * 32 + h * 8);
  f.u[1] = *reinterpret_cast<const uint4*>(pl + r * 32 + 16 + h * 8);
  return f.v;
}

__device__ inline f32x8 wmma_bf16(bf16x16 a, bf16x16 b, f32x8 c) {
  return __builtin_amdgcn_wmma_f32_16x16x32_bf16(false, a, false, b, (short)0, c, false, false);
}

// ------------------------------------------------------------------
// fp32 -> bf16 conversion
__global__ __launch_bounds__(256) void cvt_bf16_kernel(const float* __restrict__ in,
                                                       __hip_bfloat16* __restrict__ out, int n) {
  int i = blockIdx.x * 256 + threadIdx.x;
  if (i < n) out[i] = __float2bfloat16(in[i]);
}

// ------------------------------------------------------------------
// LayerNorm over C=768, one block per row; output bf16.
__global__ __launch_bounds__(256) void ln_kernel(const float* __restrict__ x,
                                                 const float* __restrict__ g,
                                                 const float* __restrict__ b,
                                                 __hip_bfloat16* __restrict__ out) {
  __shared__ float red[256];
  int row = blockIdx.x, tid = threadIdx.x;
  const float* xr = x + (size_t)row * CH;
  float v0 = xr[tid], v1 = xr[tid + 256], v2 = xr[tid + 512];
  red[tid] = v0 + v1 + v2;
  __syncthreads();
  for (int st = 128; st > 0; st >>= 1) { if (tid < st) red[tid] += red[tid + st]; __syncthreads(); }
  float mu = red[0] * (1.0f / CH);
  __syncthreads();
  float d0 = v0 - mu, d1 = v1 - mu, d2 = v2 - mu;
  red[tid] = d0 * d0 + d1 * d1 + d2 * d2;
  __syncthreads();
  for (int st = 128; st > 0; st >>= 1) { if (tid < st) red[tid] += red[tid + st]; __syncthreads(); }
  float rs = rsqrtf(red[0] * (1.0f / CH) + 1e-5f);
  __hip_bfloat16* orow = out + (size_t)row * CH;
  orow[tid]       = __float2bfloat16(d0 * rs * g[tid]       + b[tid]);
  orow[tid + 256] = __float2bfloat16(d1 * rs * g[tid + 256] + b[tid + 256]);
  orow[tid + 512] = __float2bfloat16(d2 * rs * g[tid + 512] + b[tid + 512]);
}

// ------------------------------------------------------------------
// V [B,H,N,D] -> Vt [B,H,D,N]
__global__ __launch_bounds__(256) void vtrans_kernel(const __hip_bfloat16* __restrict__ v,
                                                     __hip_bfloat16* __restrict__ vt) {
  int idx = blockIdx.x * 256 + threadIdx.x;          // over B*H*N*D (exact)
  int d = idx & (HDIM - 1);
  int t = (idx >> 6) & (SEQ - 1);
  int bh = idx >> 17;
  vt[((size_t)bh * HDIM + d) * SEQ + t] = v[idx];
}

// ------------------------------------------------------------------
// Generic bf16 WMMA GEMM: out[m,n] = epilogue( sum_k A[m,k] * Bw[n,k] )
// A: MxK row-major bf16, Bw: NxK row-major bf16 (X @ W^T pattern).
// Block: 256 threads = 8 waves, each wave owns a 32x64 tile -> block tile 128x128.
// Per K-step: 2 A-frags x 4 B-frags = 8 independent WMMAs (covers WMMA RAW latency).
enum { EPI_Q = 0, EPI_KV = 1, EPI_GELU = 2, EPI_RES = 3 };

template <int MODE>
__global__ __launch_bounds__(256) void gemm_bf16_kernel(
    const __hip_bfloat16* __restrict__ A, const __hip_bfloat16* __restrict__ Bw,
    const float* __restrict__ bias, const float* __restrict__ resid,
    void* __restrict__ out0, void* __restrict__ out1, int Ncols, int K) {
  int lane = threadIdx.x & 31;
  int wid  = threadIdx.x >> 5;
  int m0 = blockIdx.y * 128 + (wid >> 1) * 32;
  int n0 = blockIdx.x * 128 + (wid & 1) * 64;

  f32x8 c[2][4];
  for (int i = 0; i < 2; ++i)
    for (int j = 0; j < 4; ++j)
      for (int r = 0; r < 8; ++r) c[i][j][r] = 0.0f;

  for (int k = 0; k < K; k += 32) {
    bf16x16 a0 = load_frag_a(A, K, m0, k, lane);
    bf16x16 a1 = load_frag_a(A, K, m0 + 16, k, lane);
    bf16x16 b0 = load_frag_b(Bw, K, n0, k, lane);
    bf16x16 b1 = load_frag_b(Bw, K, n0 + 16, k, lane);
    bf16x16 b2 = load_frag_b(Bw, K, n0 + 32, k, lane);
    bf16x16 b3 = load_frag_b(Bw, K, n0 + 48, k, lane);
    c[0][0] = wmma_bf16(a0, b0, c[0][0]);
    c[0][1] = wmma_bf16(a0, b1, c[0][1]);
    c[0][2] = wmma_bf16(a0, b2, c[0][2]);
    c[0][3] = wmma_bf16(a0, b3, c[0][3]);
    c[1][0] = wmma_bf16(a1, b0, c[1][0]);
    c[1][1] = wmma_bf16(a1, b1, c[1][1]);
    c[1][2] = wmma_bf16(a1, b2, c[1][2]);
    c[1][3] = wmma_bf16(a1, b3, c[1][3]);
  }

  int half = lane >> 4, ln = lane & 15;
  for (int tm = 0; tm < 2; ++tm)
    for (int tn = 0; tn < 4; ++tn)
      for (int r = 0; r < 8; ++r) {
        int rowg = m0 + tm * 16 + r + half * 8;
        int col  = n0 + tn * 16 + ln;
        float acc = c[tm][tn][r];
        if (MODE == EPI_Q) {
          int bbi = rowg >> 11, tok = rowg & (SEQ - 1);
          int hh = col >> 6, dd = col & (HDIM - 1);
          ((__hip_bfloat16*)out0)[(((size_t)bbi * HEADS + hh) * SEQ + tok) * HDIM + dd] =
              __float2bfloat16(acc);
        } else if (MODE == EPI_KV) {
          int bbi = rowg >> 11, tok = rowg & (SEQ - 1);
          int cc = col;
          __hip_bfloat16* dst = (__hip_bfloat16*)out0;
          if (cc >= CH) { dst = (__hip_bfloat16*)out1; cc -= CH; }
          int hh = cc >> 6, dd = cc & (HDIM - 1);
          dst[(((size_t)bbi * HEADS + hh) * SEQ + tok) * HDIM + dd] = __float2bfloat16(acc);
        } else if (MODE == EPI_GELU) {
          float v = acc + bias[col];
          v = 0.5f * v * (1.0f + erff(v * 0.70710678118654752f));   // exact GELU
          ((__hip_bfloat16*)out0)[(size_t)rowg * Ncols + col] = __float2bfloat16(v);
        } else { // EPI_RES
          float v = acc + bias[col] + resid[(size_t)rowg * Ncols + col];
          ((float*)out0)[(size_t)rowg * Ncols + col] = v;
        }
      }
}

// ------------------------------------------------------------------
// Flash attention: one wave per (b, h, 16-query tile); online softmax.
__global__ __launch_bounds__(32) void attn_kernel(const __hip_bfloat16* __restrict__ q,
                                                  const __hip_bfloat16* __restrict__ k,
                                                  const __hip_bfloat16* __restrict__ vT,
                                                  const float* __restrict__ mask,
                                                  __hip_bfloat16* __restrict__ o) {
  __shared__ __align__(16) __hip_bfloat16 pl[16 * 32];  // P staging tile (C-layout -> A-layout)
  int lane = threadIdx.x & 31;
  int half = lane >> 4, ln = lane & 15;

  int bid = blockIdx.x;                 // = ((bb*HEADS)+hh)*(SEQ/16) + qt
  int qt = bid & 127;
  int hh = (bid >> 7) % HEADS;
  int bb = bid / (128 * HEADS);

  const __hip_bfloat16* qp = q  + ((size_t)bb * HEADS + hh) * SEQ * HDIM;
  const __hip_bfloat16* kp = k  + ((size_t)bb * HEADS + hh) * SEQ * HDIM;
  const __hip_bfloat16* vp = vT + ((size_t)bb * HEADS + hh) * HDIM * SEQ;
  const float* mp = mask + ((size_t)bb * SEQ + qt * 16) * SEQ;

  bf16x16 a0 = load_frag_a(qp, HDIM, qt * 16, 0, lane);
  bf16x16 a1 = load_frag_a(qp, HDIM, qt * 16, 32, lane);

  f32x8 acc[4];
  float mi[8], li[8];
  for (int t = 0; t < 4; ++t)
    for (int r = 0; r < 8; ++r) acc[t][r] = 0.0f;
  for (int r = 0; r < 8; ++r) { mi[r] = -1e30f; li[r] = 0.0f; }

  for (int kb = 0; kb < SEQ; kb += 32) {
    // S = (Q K^T) * scale + (-1e5 * mask)
    bf16x16 bk00 = load_frag_b(kp, HDIM, kb, 0, lane);
    bf16x16 bk01 = load_frag_b(kp, HDIM, kb, 32, lane);
    bf16x16 bk10 = load_frag_b(kp, HDIM, kb + 16, 0, lane);
    bf16x16 bk11 = load_frag_b(kp, HDIM, kb + 16, 32, lane);
    f32x8 s0, s1;
    for (int r = 0; r < 8; ++r) { s0[r] = 0.0f; s1[r] = 0.0f; }
    s0 = wmma_bf16(a0, bk00, s0);
    s0 = wmma_bf16(a1, bk01, s0);
    s1 = wmma_bf16(a0, bk10, s1);
    s1 = wmma_bf16(a1, bk11, s1);

    for (int r = 0; r < 8; ++r) {
      int rowl = r + half * 8;
      float mk0 = mp[(size_t)rowl * SEQ + kb + ln];
      float mk1 = mp[(size_t)rowl * SEQ + kb + 16 + ln];
      s0[r] = s0[r] * ATTN_SCALE - 100000.0f * mk0;
      s1[r] = s1[r] * ATTN_SCALE - 100000.0f * mk1;
    }

    // online softmax over this 32-key block
    for (int r = 0; r < 8; ++r) {
      float t = fmaxf(s0[r], s1[r]);
      for (int off = 1; off < 16; off <<= 1) t = fmaxf(t, __shfl_xor(t, off, 32));
      float mn = fmaxf(mi[r], t);
      float fsc = __expf(mi[r] - mn);
      float p0 = __expf(s0[r] - mn);
      float p1 = __expf(s1[r] - mn);
      float rs = p0 + p1;
      for (int off = 1; off < 16; off <<= 1) rs += __shfl_xor(rs, off, 32);
      li[r] = li[r] * fsc + rs;
      mi[r] = mn;
      acc[0][r] *= fsc; acc[1][r] *= fsc; acc[2][r] *= fsc; acc[3][r] *= fsc;
      int rowl = r + half * 8;
      pl[rowl * 32 + ln]      = __float2bfloat16(p0);
      pl[rowl * 32 + 16 + ln] = __float2bfloat16(p1);
    }
    __syncthreads();  // single-wave WG: S_NOP, keeps DS ordering explicit

    bf16x16 pf = load_frag_a_lds(pl, lane);
    for (int dt = 0; dt < 4; ++dt) {
      bf16x16 bv = load_frag_b(vp, SEQ, dt * 16, kb, lane);  // rows = d, K = keys
      acc[dt] = wmma_bf16(pf, bv, acc[dt]);
    }
    __syncthreads();
  }

  // normalize + write o in [B*N, C] token-major bf16
  for (int dt = 0; dt < 4; ++dt)
    for (int r = 0; r < 8; ++r) {
      int tok = qt * 16 + r + half * 8;
      int col = hh * HDIM + dt * 16 + ln;
      o[((size_t)bb * SEQ + tok) * CH + col] = __float2bfloat16(acc[dt][r] / li[r]);
    }
}

// ------------------------------------------------------------------
extern "C" void kernel_launch(void* const* d_in, const int* in_sizes, int n_in,
                              void* d_out, int out_size, void* d_ws, size_t ws_size,
                              hipStream_t stream) {
  const float* x    = (const float*)d_in[0];
  const float* mask = (const float*)d_in[1];
  const float* g1   = (const float*)d_in[2];
  const float* b1   = (const float*)d_in[3];
  const float* Wq   = (const float*)d_in[4];
  const float* Wkv  = (const float*)d_in[5];
  const float* Wp   = (const float*)d_in[6];
  const float* bp   = (const float*)d_in[7];
  const float* g2   = (const float*)d_in[8];
  const float* b2   = (const float*)d_in[9];
  const float* W1   = (const float*)d_in[10];
  const float* bf1  = (const float*)d_in[11];
  const float* W2   = (const float*)d_in[12];
  const float* bf2  = (const float*)d_in[13];

  char* cur = (char*)d_ws;
  auto alloc = [&](size_t bytes) -> char* {
    char* p = cur;
    cur += (bytes + 255) & ~(size_t)255;
    return p;
  };
  const size_t MC = (size_t)MROWS * CH;
  __hip_bfloat16* xnb  = (__hip_bfloat16*)alloc(MC * 2);                    // LN1 out
  __hip_bfloat16* wqb  = (__hip_bfloat16*)alloc((size_t)CH * CH * 2);
  __hip_bfloat16* wkvb = (__hip_bfloat16*)alloc((size_t)2 * CH * CH * 2);
  __hip_bfloat16* wpb  = (__hip_bfloat16*)alloc((size_t)CH * CH * 2);
  __hip_bfloat16* w1b  = (__hip_bfloat16*)alloc((size_t)HID_ * CH * 2);
  __hip_bfloat16* w2b  = (__hip_bfloat16*)alloc((size_t)CH * HID_ * 2);
  __hip_bfloat16* qb   = (__hip_bfloat16*)alloc(MC * 2);                    // [B,H,N,D]
  __hip_bfloat16* kbuf = (__hip_bfloat16*)alloc(MC * 2);                    // [B,H,N,D]
  __hip_bfloat16* vbuf = (__hip_bfloat16*)alloc(MC * 2);                    // [B,H,N,D]
  __hip_bfloat16* vtb  = (__hip_bfloat16*)alloc(MC * 2);                    // [B,H,D,N]
  __hip_bfloat16* ob   = (__hip_bfloat16*)alloc(MC * 2);                    // attn out, token-major
  float*          x1   = (float*)alloc(MC * 4);                             // residual 1
  __hip_bfloat16* xn2b = (__hip_bfloat16*)alloc(MC * 2);                    // LN2 out
  __hip_bfloat16* hb   = (__hip_bfloat16*)alloc((size_t)MROWS * HID_ * 2);  // GELU out
  (void)ws_size; (void)in_sizes; (void)n_in; (void)out_size;

  // 1. weights -> bf16
  cvt_bf16_kernel<<<(CH * CH + 255) / 256, 256, 0, stream>>>(Wq, wqb, CH * CH);
  cvt_bf16_kernel<<<(2 * CH * CH + 255) / 256, 256, 0, stream>>>(Wkv, wkvb, 2 * CH * CH);
  cvt_bf16_kernel<<<(CH * CH + 255) / 256, 256, 0, stream>>>(Wp, wpb, CH * CH);
  cvt_bf16_kernel<<<(HID_ * CH + 255) / 256, 256, 0, stream>>>(W1, w1b, HID_ * CH);
  cvt_bf16_kernel<<<(CH * HID_ + 255) / 256, 256, 0, stream>>>(W2, w2b, CH * HID_);

  // 2. LN1
  ln_kernel<<<MROWS, 256, 0, stream>>>(x, g1, b1, xnb);

  // 3. Q / KV projections (head-major scatter epilogues)
  gemm_bf16_kernel<EPI_Q><<<dim3(CH / 128, MROWS / 128), 256, 0, stream>>>(
      xnb, wqb, nullptr, nullptr, qb, nullptr, CH, CH);
  gemm_bf16_kernel<EPI_KV><<<dim3(2 * CH / 128, MROWS / 128), 256, 0, stream>>>(
      xnb, wkvb, nullptr, nullptr, kbuf, vbuf, 2 * CH, CH);

  // 4. V -> V^T
  vtrans_kernel<<<(BATCH * HEADS * SEQ * HDIM) / 256, 256, 0, stream>>>(vbuf, vtb);

  // 5. attention
  attn_kernel<<<BATCH * HEADS * (SEQ / 16), 32, 0, stream>>>(qb, kbuf, vtb, mask, ob);

  // 6. out-proj + residual -> x1 (fp32)
  gemm_bf16_kernel<EPI_RES><<<dim3(CH / 128, MROWS / 128), 256, 0, stream>>>(
      ob, wpb, bp, x, x1, nullptr, CH, CH);

  // 7. LN2
  ln_kernel<<<MROWS, 256, 0, stream>>>(x1, g2, b2, xn2b);

  // 8. FC1 + GELU -> hb (bf16)
  gemm_bf16_kernel<EPI_GELU><<<dim3(HID_ / 128, MROWS / 128), 256, 0, stream>>>(
      xn2b, w1b, bf1, nullptr, hb, nullptr, HID_, CH);

  // 9. FC2 + bias + residual -> d_out (fp32)
  gemm_bf16_kernel<EPI_RES><<<dim3(CH / 128, MROWS / 128), 256, 0, stream>>>(
      hb, w2b, bf2, x1, (float*)d_out, nullptr, CH, HID_);
}
